// DeformConv_23656679866428
// MI455X (gfx1250) — matrix operand order repaired and checked by reading, not compiled, not run on hardware
//
#include <hip/hip_runtime.h>
#include <hip/hip_bf16.h>
#include <math.h>

// ---------------------------------------------------------------------------
// Deformable Conv v2 for MI455X (gfx1250, wave32, WMMA)
//   B=8, CIN=COUT=64, H=W=96, 3x3, stride=1, pad=1, dil=1
// Pipeline:
//   K1: pack weight -> f16 Wmat[o][k*64+c]              (GEMM A-operand source)
//   K2: offset conv (27ch) -> dy/dx/sigmoid(mask) planes in workspace
//   K3: fused bilinear sample (masked, ->f16 LDS) + v_wmma_f32_16x16x32_f16 GEMM
// ---------------------------------------------------------------------------

typedef __attribute__((ext_vector_type(16))) _Float16 v16h;
typedef __attribute__((ext_vector_type(8)))  _Float16 v8h;
typedef __attribute__((ext_vector_type(8)))  float    v8f;

#define CIN   64
#define COUT  64
#define KTAPS 9
#define HW    96
#define PIX   (HW * HW)        // 9216
#define BATCH 8
#define CK    (KTAPS * CIN)    // 576
#define TILE_P 64              // pixels per workgroup in K3
#define LDS_STRIDE 72          // halfs per pixel row (64 + 8 pad, 16B aligned)

// ---------------------------------------------------------------------------
// K1: Wmat[o][k*64 + c] = (f16) weight[o][c][k]
// ---------------------------------------------------------------------------
__global__ __launch_bounds__(256) void pack_weight_kernel(
    const float* __restrict__ w, _Float16* __restrict__ Wmat)
{
    int i = blockIdx.x * 256 + threadIdx.x;      // 0 .. 64*576-1
    int o = i / CK;
    int r = i - o * CK;
    int k = r / CIN;
    int c = r - k * CIN;
    Wmat[i] = (_Float16)w[(o * CIN + c) * KTAPS + k];
}

// ---------------------------------------------------------------------------
// K2: offset conv. Block = one output channel o (0..26), 256 pixels.
// w_off[o][:] (576 f32) lives in LDS -> scalar-broadcast reads.
// ---------------------------------------------------------------------------
__global__ __launch_bounds__(256) void offset_conv_kernel(
    const float* __restrict__ x, const float* __restrict__ w_off,
    const float* __restrict__ b_off,
    float* __restrict__ dyA, float* __restrict__ dxA, float* __restrict__ mA)
{
    __shared__ float wsm[CK];
    const int o = blockIdx.x;                    // 0..26
    for (int i = threadIdx.x; i < CK; i += 256)
        wsm[i] = w_off[o * CK + i];
    __syncthreads();

    const int gp = blockIdx.y * 256 + threadIdx.x;   // 0 .. B*PIX-1 (exact)
    const int b  = gp / PIX;
    const int p  = gp - b * PIX;
    const int oy = p / HW;
    const int ox = p - oy * HW;
    const float* xb = x + (size_t)b * CIN * PIX;

    float acc = b_off[o];
    for (int c = 0; c < CIN; ++c) {
        const float* xc = xb + c * PIX;
        const float* wc = wsm + c * KTAPS;
        #pragma unroll
        for (int t = 0; t < KTAPS; ++t) {
            const int yy = oy + t / 3 - 1;
            const int xx = ox + t % 3 - 1;
            float xv = (yy >= 0 && yy < HW && xx >= 0 && xx < HW)
                           ? xc[yy * HW + xx] : 0.0f;
            acc = fmaf(xv, wc[t], acc);
        }
    }

    if (o < 18) {                                // interleaved (dy,dx) per tap
        const int k = o >> 1;
        const int idx = (b * KTAPS + k) * PIX + p;
        if (o & 1) dxA[idx] = acc; else dyA[idx] = acc;
    } else {                                     // mask logits -> sigmoid
        const int k = o - 18;
        mA[(b * KTAPS + k) * PIX + p] = 1.0f / (1.0f + __expf(-acc));
    }
}

// ---------------------------------------------------------------------------
// K3: fused deformable sampling + WMMA GEMM.
// Block: 128 threads = 4 waves; owns (batch b, 64-pixel tile).
// Per tap: sample 64ch x 64pix (masked, f16) into LDS[pixel][c],
//          then each wave does a 64x16 GEMM slice with 16x16x32 f16 WMMA.
// ---------------------------------------------------------------------------
__global__ __launch_bounds__(128) void deform_wmma_kernel(
    const float* __restrict__ x,
    const float* __restrict__ dyA, const float* __restrict__ dxA,
    const float* __restrict__ mA,
    const _Float16* __restrict__ Wmat, const float* __restrict__ bias,
    float* __restrict__ out)
{
    __shared__ _Float16 lds[TILE_P * LDS_STRIDE];

    const int b     = blockIdx.y;
    const int pbase = blockIdx.x * TILE_P;
    const int tid   = threadIdx.x;
    const int lane  = tid & 31;
    const int wave  = tid >> 5;
    const int n     = lane & 15;     // column within 16x16 tile
    const int sel   = lane >> 4;     // K-half / row-half selector

    // Accumulators: 4 M-tiles of 16x16 f32, seeded with bias.
    // C/D layout: lanes 0-15 VGPR r -> M=r ; lanes 16-31 VGPR r -> M=8+r.
    v8f acc[4];
    #pragma unroll
    for (int mt = 0; mt < 4; ++mt)
        #pragma unroll
        for (int r = 0; r < 8; ++r)
            acc[mt][r] = bias[mt * 16 + sel * 8 + r];

    // Sampling-phase mapping: 2 threads per pixel, 32 channels each.
    const int p_loc = tid >> 1;                 // 0..63
    const int chalf = (tid & 1) * 32;           // channel base
    const int p  = pbase + p_loc;
    const int oy = p / HW;
    const int ox = p - oy * HW;
    const float* xb = x + (size_t)b * CIN * PIX;

    for (int k = 0; k < KTAPS; ++k) {
        // ---------------- bilinear sampling into LDS ----------------
        const int off = (b * KTAPS + k) * PIX + p;
        const float dy = dyA[off];
        const float dx = dxA[off];
        const float m  = mA[off];

        const float ys = (float)(oy - 1 + k / 3) + dy;
        const float xs = (float)(ox - 1 + k % 3) + dx;
        const float y0f = floorf(ys), x0f = floorf(xs);
        const float wy1 = ys - y0f,   wx1 = xs - x0f;
        const float wy0 = 1.0f - wy1, wx0 = 1.0f - wx1;
        const int y0 = (int)y0f, x0 = (int)x0f;
        const int y1 = y0 + 1,   x1 = x0 + 1;

        int   i00, i01, i10, i11;
        float w00, w01, w10, w11;
        {
            const int y0c = min(max(y0, 0), HW - 1);
            const int y1c = min(max(y1, 0), HW - 1);
            const int x0c = min(max(x0, 0), HW - 1);
            const int x1c = min(max(x1, 0), HW - 1);
            const bool vy0 = (y0 >= 0) & (y0 < HW);
            const bool vy1 = (y1 >= 0) & (y1 < HW);
            const bool vx0 = (x0 >= 0) & (x0 < HW);
            const bool vx1 = (x1 >= 0) & (x1 < HW);
            i00 = y0c * HW + x0c;  w00 = (vy0 & vx0) ? wy0 * wx0 : 0.0f;
            i01 = y0c * HW + x1c;  w01 = (vy0 & vx1) ? wy0 * wx1 : 0.0f;
            i10 = y1c * HW + x0c;  w10 = (vy1 & vx0) ? wy1 * wx0 : 0.0f;
            i11 = y1c * HW + x1c;  w11 = (vy1 & vx1) ? wy1 * wx1 : 0.0f;
        }

        _Float16 buf[32];
        const float* xc = xb + (size_t)chalf * PIX;
        #pragma unroll 4
        for (int c = 0; c < 32; ++c) {
            float v = xc[i00] * w00 + xc[i01] * w01
                    + xc[i10] * w10 + xc[i11] * w11;   // L2-resident gathers
            buf[c] = (_Float16)(v * m);
            xc += PIX;
        }
        _Float16* dst = &lds[p_loc * LDS_STRIDE + chalf];
        #pragma unroll
        for (int j = 0; j < 4; ++j)
            *(v8h*)(dst + j * 8) = *(const v8h*)(buf + j * 8);  // ds_store_b128

        __syncthreads();

        // ---------------- WMMA GEMM: wave -> 64(out) x 16(pix) ----------------
        // B frag (32x16 f16): lanes 0-15 K=0..15, lanes 16-31 K=16..31,
        //   column = pixel -> one contiguous 32B run of LDS row.
        const _Float16* ldsB = &lds[(wave * 16 + n) * LDS_STRIDE];
        #pragma unroll
        for (int chunk = 0; chunk < 2; ++chunk) {
            const int ckb = k * CIN + chunk * 32;
            const _Float16* bp = ldsB + chunk * 32 + sel * 16;
            v16h bfrag;
            ((v8h*)&bfrag)[0] = *(const v8h*)(bp);       // ds_load_b128
            ((v8h*)&bfrag)[1] = *(const v8h*)(bp + 8);   // ds_load_b128
            #pragma unroll
            for (int mt = 0; mt < 4; ++mt) {
                // A frag (16x32 f16): lane<16 K{0-15}, lane>=16 K{16-31}
                const _Float16* ap = Wmat + (mt * 16 + n) * CK + ckb;
                v16h afrag;
                ((v8h*)&afrag)[0] = *(const v8h*)(ap + sel * 16);
                ((v8h*)&afrag)[1] = *(const v8h*)(ap + sel * 16 + 8);
                acc[mt] = __builtin_amdgcn_wmma_f32_16x16x32_f16(
                    false, afrag, false, bfrag, (short)0, acc[mt], false, false);
            }
        }
        __syncthreads();   // protect LDS before next tap overwrites it
    }

    // ---------------- write out ----------------
    const int pcol = pbase + wave * 16 + n;
    #pragma unroll
    for (int mt = 0; mt < 4; ++mt) {
        #pragma unroll
        for (int r = 0; r < 8; ++r) {
            const int o = mt * 16 + sel * 8 + r;
            out[((size_t)(b * COUT + o)) * PIX + pcol] = acc[mt][r];
        }
    }
}

// ---------------------------------------------------------------------------
// Host launcher
// ---------------------------------------------------------------------------
extern "C" void kernel_launch(void* const* d_in, const int* in_sizes, int n_in,
                              void* d_out, int out_size, void* d_ws, size_t ws_size,
                              hipStream_t stream)
{
    (void)in_sizes; (void)n_in; (void)out_size; (void)ws_size;

    const float* x      = (const float*)d_in[0];   // (8,64,96,96)
    const float* w_off  = (const float*)d_in[1];   // (27,64,3,3)
    const float* b_off  = (const float*)d_in[2];   // (27,)
    const float* weight = (const float*)d_in[3];   // (64,64,3,3)
    const float* bias   = (const float*)d_in[4];   // (64,)
    float*       out    = (float*)d_out;           // (8,64,96,96)

    // Workspace carve-up (all 256B aligned).
    const size_t plane = (size_t)BATCH * KTAPS * PIX;   // 663552 floats
    float*     dyA  = (float*)d_ws;
    float*     dxA  = dyA + plane;
    float*     mA   = dxA + plane;
    _Float16*  Wmat = (_Float16*)(mA + plane);          // 64*576 halfs

    // K1: pack weights to f16, tap-major K ordering.
    pack_weight_kernel<<<dim3((COUT * CK) / 256), dim3(256), 0, stream>>>(weight, Wmat);

    // K2: offset conv -> dy/dx/mask planes. grid = (27 channels, 73728/256 pixels)
    offset_conv_kernel<<<dim3(27, (BATCH * PIX) / 256), dim3(256), 0, stream>>>(
        x, w_off, b_off, dyA, dxA, mA);

    // K3: fused sample + WMMA GEMM. grid = (9216/64 tiles, 8 batches), 4 waves.
    deform_wmma_kernel<<<dim3(PIX / TILE_P, BATCH), dim3(128), 0, stream>>>(
        x, dyA, dxA, mA, Wmat, bias, out);
}
